// FrequencyLinear_37589553774901
// MI455X (gfx1250) — matrix operand order repaired and checked by reading, not compiled, not run on hardware
//
#include <hip/hip_runtime.h>

// ---- types ----
typedef __attribute__((ext_vector_type(16))) __bf16         v16bf;
typedef __attribute__((ext_vector_type(8)))  float          v8f;
typedef __attribute__((ext_vector_type(4)))  float          f4;
typedef __attribute__((ext_vector_type(8)))  unsigned short v8u;
typedef __attribute__((ext_vector_type(16))) unsigned short v16u;
typedef __attribute__((ext_vector_type(4)))  unsigned short u4;

#define M_DIM 4096   // batch
#define N_DIM 2048   // out_features
#define K_DIM 2048   // in_features
#define BM 128
#define BN 128
#define KC 32
#define LDSS 40      // KC + 8 pad elements (80B row stride, 16B aligned)

// fp32 -> bf16 round-to-nearest-even
__device__ __forceinline__ unsigned short f2bf(float f) {
  unsigned int u = __builtin_bit_cast(unsigned int, f);
  u += 0x7FFFu + ((u >> 16) & 1u);
  return (unsigned short)(u >> 16);
}

__device__ __forceinline__ void cvt_store4(unsigned short* dst, f4 v) {
  u4 r;
  r[0] = f2bf(v[0]); r[1] = f2bf(v[1]); r[2] = f2bf(v[2]); r[3] = f2bf(v[3]);
  *(u4*)dst = r;  // 8-byte LDS store
}

// two 16B LDS loads -> one v16bf fragment
__device__ __forceinline__ v16bf load_frag(const unsigned short* p) {
  v8u lo = *(const v8u*)p;
  v8u hi = *(const v8u*)(p + 8);
  v16u r = __builtin_shufflevector(lo, hi, 0,1,2,3,4,5,6,7,8,9,10,11,12,13,14,15);
  return __builtin_bit_cast(v16bf, r);
}

__device__ __forceinline__ v8f wmma_bf16(v16bf a, v16bf b, v8f c) {
  return __builtin_amdgcn_wmma_f32_16x16x32_bf16(
      /*neg_a=*/false, a, /*neg_b=*/false, b,
      /*c_mod=*/(short)0, c, /*reuse_a=*/false, /*reuse_b=*/false);
}

__global__ __launch_bounds__(256)
void FrequencyLinear_cgemm_wmma(const float* __restrict__ x_re,
                                const float* __restrict__ x_im,
                                const float* __restrict__ w_re,
                                const float* __restrict__ w_im,
                                float* __restrict__ out) {
  __shared__ __align__(16) unsigned short sA[2][BM][LDSS]; // [re/im][row][k]
  __shared__ __align__(16) unsigned short sB[2][BN][LDSS]; // [re/im][row][k]

  const int t    = threadIdx.x;
  const int lane = t & 31;
  const int wid  = t >> 5;
  const int m_off = (wid & 3) * 32;   // 4 waves along M
  const int n_off = (wid >> 2) * 64;  // 2 waves along N
  const int bm = blockIdx.y * BM;
  const int bn = blockIdx.x * BN;

  const int lr  = lane & 15;          // row within 16x16 tile
  const int g32 = (lane >> 4) * 16;   // fragment element base (per lane half)

  v8f acc_re[2][4], acc_im[2][4];
#pragma unroll
  for (int mi = 0; mi < 2; ++mi)
#pragma unroll
    for (int ni = 0; ni < 4; ++ni) { acc_re[mi][ni] = (v8f)(0.0f); acc_im[mi][ni] = (v8f)(0.0f); }

  const v16u sgn = {0x8000,0x8000,0x8000,0x8000,0x8000,0x8000,0x8000,0x8000,
                    0x8000,0x8000,0x8000,0x8000,0x8000,0x8000,0x8000,0x8000};

  for (int k0 = 0; k0 < K_DIM; k0 += KC) {
    // ---- stage global fp32 -> LDS bf16 (256 threads cover 128x32 x4 matrices) ----
#pragma unroll
    for (int i = 0; i < 4; ++i) {
      const int linear = t + i * 256;        // 0..1023
      const int row = linear >> 3;           // 0..127
      const int kq  = (linear & 7) * 4;      // 0,4,...,28
      const int b   = kq >> 3;               // 8-wide block index 0..3
      const int pA  = ((((b & 1) << 1) | (b >> 1)) << 3) + (kq & 7); // A frag K-permute
      const size_t ga = (size_t)(bm + row) * K_DIM + k0 + kq;
      const size_t gw = (size_t)(bn + row) * K_DIM + k0 + kq;
      cvt_store4(&sA[0][row][kq], *(const f4*)(x_re + ga));
      cvt_store4(&sA[1][row][kq], *(const f4*)(x_im + ga));
      cvt_store4(&sB[0][row][kq], *(const f4*)(w_re + gw));
      cvt_store4(&sB[1][row][kq], *(const f4*)(w_im + gw));
      if (k0 + KC < K_DIM) {
        __builtin_prefetch(x_re + ga + KC, 0, 0);
        __builtin_prefetch(w_re + gw + KC, 0, 0);
      }
    }
    __syncthreads();

    // ---- A fragments (shared across all 4 N tiles of this wave) ----
    v16bf a_re[2], a_im[2], a_imn[2];
#pragma unroll
    for (int mi = 0; mi < 2; ++mi) {
      const int r = m_off + mi * 16 + lr;
      a_re[mi]  = load_frag(&sA[0][r][g32]);
      a_im[mi]  = load_frag(&sA[1][r][g32]);
      a_imn[mi] = __builtin_bit_cast(v16bf, __builtin_bit_cast(v16u, a_im[mi]) ^ sgn);
    }

    // ---- B fragments + 4 complex-fused WMMAs per tile ----
#pragma unroll
    for (int ni = 0; ni < 4; ++ni) {
      const int r = n_off + ni * 16 + lr;
      v16bf b_re = load_frag(&sB[0][r][g32]);
      v16bf b_im = load_frag(&sB[1][r][g32]);
#pragma unroll
      for (int mi = 0; mi < 2; ++mi) {
        acc_re[mi][ni] = wmma_bf16(a_re[mi],  b_re, acc_re[mi][ni]);
        acc_re[mi][ni] = wmma_bf16(a_imn[mi], b_im, acc_re[mi][ni]);
        acc_im[mi][ni] = wmma_bf16(a_re[mi],  b_im, acc_im[mi][ni]);
        acc_im[mi][ni] = wmma_bf16(a_im[mi],  b_re, acc_im[mi][ni]);
      }
    }
    __syncthreads();
  }

  // ---- epilogue: D layout = 8 VGPRs, lane<16 -> M=v, lane>=16 -> M=v+8, N=lane&15 ----
  float* __restrict__ out_re = out;
  float* __restrict__ out_im = out + (size_t)M_DIM * N_DIM;
  const int gm0 = bm + m_off + ((lane >> 4) << 3);
  const int gn0 = bn + n_off + (lane & 15);
#pragma unroll
  for (int mi = 0; mi < 2; ++mi)
#pragma unroll
    for (int ni = 0; ni < 4; ++ni)
#pragma unroll
      for (int v = 0; v < 8; ++v) {
        const size_t idx = (size_t)(gm0 + mi * 16 + v) * N_DIM + (gn0 + ni * 16);
        out_re[idx] = acc_re[mi][ni][v];
        out_im[idx] = acc_im[mi][ni][v];
      }
}

extern "C" void kernel_launch(void* const* d_in, const int* in_sizes, int n_in,
                              void* d_out, int out_size, void* d_ws, size_t ws_size,
                              hipStream_t stream) {
  const float* x_re = (const float*)d_in[0];
  const float* x_im = (const float*)d_in[1];
  const float* w_re = (const float*)d_in[2];
  const float* w_im = (const float*)d_in[3];
  float* out = (float*)d_out;

  dim3 grid(N_DIM / BN, M_DIM / BM);  // 16 x 32 = 512 workgroups
  FrequencyLinear_cgemm_wmma<<<grid, 256, 0, stream>>>(x_re, x_im, w_re, w_im, out);
}
